// MuArcLayer_13623636263000
// MI455X (gfx1250) — matrix operand order repaired and checked by reading, not compiled, not run on hardware
//
#include <hip/hip_runtime.h>
#include <hip/hip_bf16.h>
#include <math.h>

// ---------------------------------------------------------------------------
// Types for WMMA (gfx1250, wave32)
// ---------------------------------------------------------------------------
typedef __attribute__((ext_vector_type(16))) _Float16 v16h;
typedef __attribute__((ext_vector_type(8)))  _Float16 v8h;
typedef __attribute__((ext_vector_type(8)))  float    v8f;

union HFrag { v16h v; v8h h[2]; };

// Low 32 bits of a generic pointer to a __shared__ object == LDS byte offset
// (generic LDS addr = {SHARED_BASE[63:32], offset[31:0]}, ISA 10.2 aperture rules).
__device__ __forceinline__ unsigned lds_off(const void* p) {
  return (unsigned)(unsigned long long)p;
}

// ---------------------------------------------------------------------------
// Tiled f16 WMMA GEMM with f32 accumulation.
//   C[M,N] = sum_k A[m,k] * Bt[n,k]        (both operands K-contiguous)
// BM=BN=128, BK=64: 16 v_wmma_f32_16x16x32_f16 per sync point.
// Tile staging uses GLOBAL_LOAD_ASYNC_TO_LDS_B128 (ASYNCcnt-tracked DMA,
// no VGPR staging) with double-buffered LDS tiles (72 KB of the WGP's 320 KB).
// MODE 0: store f16 (Ch)
// MODE 1: store f32 (Cf)
// MODE 2: store f32 = acc + resid
// MODE 3: store f16 = gelu(acc + bias[n])      (exact erf GELU)
// MODE 4: store f32 = acc + bias[n] + resid
// Batched via gridDim.z with element strides sA/sB/sC/sR (0 => shared).
// Requires M,N multiples of 128 and K multiple of 64.
// ---------------------------------------------------------------------------
template <int MODE>
__global__ __launch_bounds__(256) void gemm_wmma_kernel(
    const _Float16* __restrict__ A, const _Float16* __restrict__ Bt,
    int M, int N, int K,
    long long sA, long long sB, long long sC, long long sR,
    const float* __restrict__ bias, const float* __restrict__ resid,
    float* __restrict__ Cf, _Float16* __restrict__ Ch)
{
  __shared__ __align__(16) _Float16 As[2][128][72];   // 36 KB
  __shared__ __align__(16) _Float16 Bs[2][128][72];   // 36 KB

  const int tid  = threadIdx.x;
  const int lane = tid & 31;
  const int wave = tid >> 5;
  const int wm   = (wave & 1) * 64;   // wave's M offset within tile (2 waves along M)
  const int wn   = (wave >> 1) * 32;  // wave's N offset within tile (4 waves along N)

  const long long bz = blockIdx.z;
  const _Float16* Ab = A  + bz * sA;
  const _Float16* Bb = Bt + bz * sB;
  const int m_blk = blockIdx.y * 128;
  const int n_blk = blockIdx.x * 128;

  v8f acc[4][2];
  v8f zero = {0.f, 0.f, 0.f, 0.f, 0.f, 0.f, 0.f, 0.f};
#pragma unroll
  for (int i = 0; i < 4; ++i)
#pragma unroll
    for (int j = 0; j < 2; ++j) acc[i][j] = zero;

  // ISA 7.12.2 fragment gather offsets (wave32):
  //   A 16x32 f16:  lane m=l&15; halves 0..7 = K[kbA..kbA+7], 8..15 = K[kbA+16..kbA+23], kbA=(l>>4)*8
  //   B 32x16 f16:  lane n=l&15; halves 0..7 = K[kbB..kbB+7], 8..15 = K[kbB+16..kbB+23], kbB=(l>>4)*8
  const int kbA = (lane >> 4) * 8;
  const int kbB = (lane >> 4) * 8;
  const int mr  = lane & 15;

  const int tiles = K >> 6;   // BK = 64

  // Async DMA of tile `t` into LDS buffer `buf`.
  // Tile = 128 rows x 64 halves = 1024 x 16B vectors; 4 per thread per matrix
  // => 8 GLOBAL_LOAD_ASYNC_TO_LDS_B128 per thread per tile.
  auto prefetch = [&](int t, int buf) {
    const int k0 = t << 6;
#pragma unroll
    for (int u = 0; u < 4; ++u) {
      const int v   = tid + u * 256;      // 0..1023
      const int row = v >> 3;
      const int col = (v & 7) * 8;
      const unsigned la = lds_off(&As[buf][row][col]);
      const unsigned lb = lds_off(&Bs[buf][row][col]);
      const unsigned long long ga =
          (unsigned long long)(Ab + (long long)(m_blk + row) * K + k0 + col);
      const unsigned long long gb =
          (unsigned long long)(Bb + (long long)(n_blk + row) * K + k0 + col);
      asm volatile("global_load_async_to_lds_b128 %0, %1, off" :: "v"(la), "v"(ga) : "memory");
      asm volatile("global_load_async_to_lds_b128 %0, %1, off" :: "v"(lb), "v"(gb) : "memory");
    }
  };

  // 16 WMMAs on tile `buf` (two K=32 substeps).
  auto compute = [&](int buf) {
#pragma unroll
    for (int ks = 0; ks < 64; ks += 32) {
      HFrag bf[2];
#pragma unroll
      for (int j = 0; j < 2; ++j) {
        const _Float16* p = &Bs[buf][wn + j * 16 + mr][ks];
        bf[j].h[0] = *(const v8h*)(p + kbB);
        bf[j].h[1] = *(const v8h*)(p + kbB + 16);
      }
#pragma unroll
      for (int i = 0; i < 4; ++i) {
        HFrag af;
        const _Float16* p = &As[buf][wm + i * 16 + mr][ks];
        af.h[0] = *(const v8h*)(p + kbA);
        af.h[1] = *(const v8h*)(p + kbA + 16);
#pragma unroll
        for (int j = 0; j < 2; ++j)
          acc[i][j] = __builtin_amdgcn_wmma_f32_16x16x32_f16(
              false, af.v, false, bf[j].v, (short)0, acc[i][j], false, false);
      }
    }
  };

  prefetch(0, 0);

  // Steady state: branch-free (final iteration peeled).
  for (int t = 0; t < tiles - 1; ++t) {
    prefetch(t + 1, (t + 1) & 1);
    // 16 outstanding -> wait until only the newest 8 remain: tile t landed.
    asm volatile("s_wait_asynccnt 0x8" ::: "memory");
    __syncthreads();          // all waves' portions of tile t are in LDS
    compute(t & 1);
    __syncthreads();          // all reads of buf done before it is re-DMA'd
  }

  asm volatile("s_wait_asynccnt 0x0" ::: "memory");
  __syncthreads();
  compute((tiles - 1) & 1);

  // Epilogue: C/D layout — VGPR r: lanes 0-15 M=r, lanes 16-31 M=8+r; N = lane&15.
  const long long cb  = bz * sC;
  const long long rb  = bz * sR;
  const int rhi = (lane >> 4) * 8;
  const int cn  = lane & 15;
#pragma unroll
  for (int i = 0; i < 4; ++i) {
#pragma unroll
    for (int j = 0; j < 2; ++j) {
      const int gn = n_blk + wn + j * 16 + cn;
#pragma unroll
      for (int r = 0; r < 8; ++r) {
        const long long gm   = m_blk + wm + i * 16 + rhi + r;
        const long long cidx = gm * (long long)N + gn;
        const float val = acc[i][j][r];
        if (MODE == 0) {
          Ch[cb + cidx] = (_Float16)val;
        } else if (MODE == 1) {
          Cf[cb + cidx] = val;
        } else if (MODE == 2) {
          Cf[cb + cidx] = val + resid[rb + cidx];
        } else if (MODE == 3) {
          float t2 = val + bias[gn];
          Ch[cb + cidx] = (_Float16)(0.5f * t2 * (1.f + erff(t2 * 0.70710678118654752f)));
        } else { // MODE == 4
          Cf[cb + cidx] = val + bias[gn] + resid[rb + cidx];
        }
      }
    }
  }
}

// ---------------------------------------------------------------------------
// LayerNorm over last dim (D=1024), one block of 256 threads per row.
// Optionally writes f32 result, f16 result, and transposed f16 (b,d,t).
// ---------------------------------------------------------------------------
__global__ __launch_bounds__(256) void ln_kernel(
    const float* __restrict__ x, const float* __restrict__ gamma,
    const float* __restrict__ beta,
    float* __restrict__ hn, _Float16* __restrict__ h16,
    _Float16* __restrict__ hT16, int T, int D)
{
  __shared__ float red[256];
  const int row = blockIdx.x;
  const int tid = threadIdx.x;
  const float* xr = x + (size_t)row * D;

  float v[4];
  float s = 0.f;
#pragma unroll
  for (int i = 0; i < 4; ++i) { v[i] = xr[tid + i * 256]; s += v[i]; }
  red[tid] = s; __syncthreads();
  for (int st = 128; st > 0; st >>= 1) { if (tid < st) red[tid] += red[tid + st]; __syncthreads(); }
  const float mu = red[0] / (float)D; __syncthreads();

  float ss = 0.f;
#pragma unroll
  for (int i = 0; i < 4; ++i) { float d0 = v[i] - mu; ss += d0 * d0; }
  red[tid] = ss; __syncthreads();
  for (int st = 128; st > 0; st >>= 1) { if (tid < st) red[tid] += red[tid + st]; __syncthreads(); }
  const float rstd = rsqrtf(red[0] / (float)D + 1e-5f);

  const int b = row / T, tr = row % T;
#pragma unroll
  for (int i = 0; i < 4; ++i) {
    const int c = tid + i * 256;
    const float y = (v[i] - mu) * rstd * gamma[c] + beta[c];
    if (hn)  hn[(size_t)row * D + c] = y;
    h16[(size_t)row * D + c] = (_Float16)y;
    if (hT16) hT16[((size_t)b * D + c) * T + tr] = (_Float16)y;
  }
}

// q[row] = sum_c xg16[row,c] * hn[row,c]   (diagonal quadratic form)
__global__ __launch_bounds__(256) void q_kernel(
    const _Float16* __restrict__ xg, const float* __restrict__ hn,
    float* __restrict__ q, int D)
{
  __shared__ float red[256];
  const int row = blockIdx.x;
  const int tid = threadIdx.x;
  const size_t base = (size_t)row * D;
  float s = 0.f;
#pragma unroll
  for (int i = 0; i < 4; ++i) {
    const int c = tid + i * 256;
    s += (float)xg[base + c] * hn[base + c];
  }
  red[tid] = s; __syncthreads();
  for (int st = 128; st > 0; st >>= 1) { if (tid < st) red[tid] += red[tid + st]; __syncthreads(); }
  if (tid == 0) q[row] = red[0];
}

// softmax over j of s_j = 2*cross[row,j] - q_i - q_j  (row = b*T + i), write f16
__global__ __launch_bounds__(256) void softmax_kernel(
    const float* __restrict__ cross, const float* __restrict__ q,
    _Float16* __restrict__ attn, int T)
{
  __shared__ float red[256];
  const int row = blockIdx.x;
  const int tid = threadIdx.x;
  const int b = row / T, i = row % T;
  const float* cr = cross + (size_t)row * T;   // (b,i,j) => row*T + j
  const float* qb = q + (size_t)b * T;
  const float qi = qb[i];

  float sv[8];
  float mx = -3.4e38f;
#pragma unroll
  for (int k = 0; k < 8; ++k) {
    const int j = tid + k * 256;
    const float s = 2.f * cr[j] - qi - qb[j];
    sv[k] = s;
    mx = fmaxf(mx, s);
  }
  red[tid] = mx; __syncthreads();
  for (int st = 128; st > 0; st >>= 1) { if (tid < st) red[tid] = fmaxf(red[tid], red[tid + st]); __syncthreads(); }
  mx = red[0]; __syncthreads();

  float sum = 0.f;
#pragma unroll
  for (int k = 0; k < 8; ++k) { sv[k] = expf(sv[k] - mx); sum += sv[k]; }
  red[tid] = sum; __syncthreads();
  for (int st = 128; st > 0; st >>= 1) { if (tid < st) red[tid] += red[tid + st]; __syncthreads(); }
  const float inv = 1.f / red[0];
#pragma unroll
  for (int k = 0; k < 8; ++k)
    attn[(size_t)row * T + tid + k * 256] = (_Float16)(sv[k] * inv);
}

// G = I + L L^T (symmetric, 1024x1024, r=32), stored directly as f16 (N,K)-layout.
__global__ __launch_bounds__(256) void metric_kernel(
    const float* __restrict__ L, _Float16* __restrict__ g16)
{
  const int idx = blockIdx.x * 256 + threadIdx.x;
  const int n = idx >> 10;
  const int k = idx & 1023;
  float s = (n == k) ? 1.f : 0.f;
#pragma unroll
  for (int r = 0; r < 32; ++r) s += L[n * 32 + r] * L[k * 32 + r];
  g16[idx] = (_Float16)s;
}

// Wt[c*rows + r] = (f16) W[r*cols + c]   -- transpose + f32->f16
__global__ __launch_bounds__(256) void transpose_h_kernel(
    const float* __restrict__ W, _Float16* __restrict__ Wt, int rows, int cols)
{
  const size_t idx = (size_t)blockIdx.x * 256 + threadIdx.x;
  const size_t total = (size_t)rows * cols;
  if (idx >= total) return;
  const size_t c = idx / rows;
  const size_t r = idx % rows;
  Wt[idx] = (_Float16)W[r * (size_t)cols + c];
}

// ---------------------------------------------------------------------------
// Host-side orchestration (graph-capture safe: only kernel launches on stream)
// Workspace layout: ~242 MB total.
// ---------------------------------------------------------------------------
extern "C" void kernel_launch(void* const* d_in, const int* in_sizes, int n_in,
                              void* d_out, int out_size, void* d_ws, size_t ws_size,
                              hipStream_t stream) {
  (void)in_sizes; (void)n_in; (void)out_size; (void)ws_size;

  const float* x      = (const float*)d_in[0];
  const float* L      = (const float*)d_in[1];
  const float* W1     = (const float*)d_in[2];
  const float* b1     = (const float*)d_in[3];
  const float* W2     = (const float*)d_in[4];
  const float* b2     = (const float*)d_in[5];
  const float* gamma1 = (const float*)d_in[6];
  const float* beta1  = (const float*)d_in[7];
  const float* gamma2 = (const float*)d_in[8];
  const float* beta2  = (const float*)d_in[9];
  float* out = (float*)d_out;

  const int B = 4, T = 2048, D = 1024, H = 4096;
  const long long BT = (long long)B * T;   // 8192

  char* ws = (char*)d_ws;
  size_t o = 0;
  auto alloc = [&](size_t bytes) -> char* {
    char* p = ws + o;
    o += (bytes + 255) & ~(size_t)255;
    return p;
  };
  float*     hn     = (float*)    alloc((size_t)BT * D * 4);   // 32 MB
  float*     x1     = (float*)    alloc((size_t)BT * D * 4);   // 32 MB
  float*     q      = (float*)    alloc((size_t)BT * 4);       // 32 KB
  _Float16*  hn16   = (_Float16*) alloc((size_t)BT * D * 2);   // 16 MB
  _Float16*  hnT16  = (_Float16*) alloc((size_t)B * D * T * 2);// 16 MB (b,d,t)
  _Float16*  xg16   = (_Float16*) alloc((size_t)BT * D * 2);   // 16 MB
  _Float16*  hm16   = (_Float16*) alloc((size_t)BT * D * 2);   // 16 MB
  _Float16*  g16    = (_Float16*) alloc((size_t)D * D * 2);    // 2 MB
  _Float16*  w1t    = (_Float16*) alloc((size_t)H * D * 2);    // 8 MB (h,d)
  _Float16*  w2t    = (_Float16*) alloc((size_t)D * H * 2);    // 8 MB (d,h)
  _Float16*  attn16 = (_Float16*) alloc((size_t)B * T * T * 2);// 32 MB
  char*      big    = alloc((size_t)B * T * T * 4);            // 64 MB, time-shared
  float*     scores = (float*)big;        // phase 1: raw cross (b,t,t) f32
  _Float16*  hid16  = (_Float16*)big;     // phase 2: MLP hidden (bt,h) f16

  // --- precompute f16 operands ---
  metric_kernel<<<(D * D) / 256, 256, 0, stream>>>(L, g16);
  transpose_h_kernel<<<(D * H + 255) / 256, 256, 0, stream>>>(W1, w1t, D, H);
  transpose_h_kernel<<<(D * H + 255) / 256, 256, 0, stream>>>(W2, w2t, H, D);

  // --- norm1 ---
  ln_kernel<<<(int)BT, 256, 0, stream>>>(x, gamma1, beta1, hn, hn16, hnT16, T, D);

  // --- xg = hn @ G  (G symmetric => already (N,K)) ---
  gemm_wmma_kernel<0><<<dim3(D / 128, (int)BT / 128, 1), 256, 0, stream>>>(
      hn16, g16, (int)BT, D, D, 0, 0, 0, 0, nullptr, nullptr, nullptr, xg16);

  // --- q = rowdot(xg, hn) ---
  q_kernel<<<(int)BT, 256, 0, stream>>>(xg16, hn, q, D);

  // --- cross[b,i,j] = sum_d xg[b,i,d] * hn[b,j,d]  (batched) ---
  gemm_wmma_kernel<1><<<dim3(T / 128, T / 128, B), 256, 0, stream>>>(
      xg16, hn16, T, T, D,
      (long long)T * D, (long long)T * D, (long long)T * T, 0,
      nullptr, nullptr, scores, nullptr);

  // --- attn = softmax(2*cross - q_i - q_j) ---
  softmax_kernel<<<B * T, 256, 0, stream>>>(scores, q, attn16, T);

  // --- x1 = x + attn @ hn  (Bt = hn^T, batched) ---
  gemm_wmma_kernel<2><<<dim3(D / 128, T / 128, B), 256, 0, stream>>>(
      attn16, hnT16, T, D, T,
      (long long)T * T, (long long)D * T, (long long)T * D, (long long)T * D,
      nullptr, x, x1, nullptr);

  // --- norm2 ---
  ln_kernel<<<(int)BT, 256, 0, stream>>>(x1, gamma2, beta2, nullptr, hm16, nullptr, T, D);

  // --- hid = gelu(hm @ W1 + b1) ---
  gemm_wmma_kernel<3><<<dim3(H / 128, (int)BT / 128, 1), 256, 0, stream>>>(
      hm16, w1t, (int)BT, H, D, 0, 0, 0, 0, b1, nullptr, nullptr, hid16);

  // --- out = x1 + hid @ W2 + b2 ---
  gemm_wmma_kernel<4><<<dim3(D / 128, (int)BT / 128, 1), 256, 0, stream>>>(
      hid16, w2t, (int)BT, D, H, 0, 0, 0, 0, b2, x1, out, nullptr);
}
